// Block_9457517985872
// MI455X (gfx1250) — matrix operand order repaired and checked by reading, not compiled, not run on hardware
//
#include <hip/hip_runtime.h>
#include <stdint.h>

// ---------------------------------------------------------------------------
// MI455X (gfx1250) implementation: bf16 WMMA GEMMs + TDM weight-tile loads.
// Shapes: B=4, N=4096, C=1024, H=16, d=64, HIDDEN=4096, M_total = 16384 rows.
// ---------------------------------------------------------------------------

typedef __attribute__((ext_vector_type(16))) __bf16 v16bf;
typedef __attribute__((ext_vector_type(8)))  float  v8f;
typedef __attribute__((ext_vector_type(4)))  unsigned int v4u;
typedef __attribute__((ext_vector_type(8)))  int v8i;
typedef __attribute__((ext_vector_type(4)))  int v4i;

#define USE_TDM 1

__device__ __forceinline__ unsigned short f2bf(float f) {
  unsigned u = __builtin_bit_cast(unsigned, f);
  unsigned r = u + 0x7FFFu + ((u >> 16) & 1u);   // round-to-nearest-even
  return (unsigned short)(r >> 16);
}

// ---------------------------------------------------------------------------
// fp32 -> bf16 convert (weights)
// ---------------------------------------------------------------------------
__global__ __launch_bounds__(256) void f32_to_bf16_kernel(
    const float* __restrict__ in, unsigned short* __restrict__ out, int n) {
  int idx = (blockIdx.x * 256 + threadIdx.x) * 4;
  if (idx + 3 < n) {
    float4 v = *(const float4*)(in + idx);
    union { unsigned short s[4]; uint2 u; } p;
    p.s[0] = f2bf(v.x); p.s[1] = f2bf(v.y); p.s[2] = f2bf(v.z); p.s[3] = f2bf(v.w);
    *(uint2*)(out + idx) = p.u;
  }
}

// ---------------------------------------------------------------------------
// LayerNorm (C=1024) -> bf16.  One block (256 thr) per row; 4 floats/thread.
// ---------------------------------------------------------------------------
__global__ __launch_bounds__(256) void ln_to_bf16_kernel(
    const float* __restrict__ X, const float* __restrict__ gamma,
    const float* __restrict__ beta, unsigned short* __restrict__ out) {
  const int row = blockIdx.x;
  const int tid = threadIdx.x;
  const float* xr = X + (size_t)row * 1024;
  float4 xv = ((const float4*)xr)[tid];
  float s  = xv.x + xv.y + xv.z + xv.w;
  float ss = xv.x * xv.x + xv.y * xv.y + xv.z * xv.z + xv.w * xv.w;
  __shared__ float r1[256], r2[256];
  r1[tid] = s; r2[tid] = ss;
  __syncthreads();
  for (int st = 128; st > 0; st >>= 1) {
    if (tid < st) { r1[tid] += r1[tid + st]; r2[tid] += r2[tid + st]; }
    __syncthreads();
  }
  float mu   = r1[0] * (1.0f / 1024.0f);
  float var  = r2[0] * (1.0f / 1024.0f) - mu * mu;
  float rstd = rsqrtf(var + 1e-5f);
  float4 g = ((const float4*)gamma)[tid];
  float4 b = ((const float4*)beta)[tid];
  union { unsigned short s[4]; uint2 u; } p;
  p.s[0] = f2bf((xv.x - mu) * rstd * g.x + b.x);
  p.s[1] = f2bf((xv.y - mu) * rstd * g.y + b.y);
  p.s[2] = f2bf((xv.z - mu) * rstd * g.z + b.z);
  p.s[3] = f2bf((xv.w - mu) * rstd * g.w + b.w);
  ((uint2*)(out + (size_t)row * 1024))[tid] = p.u;
}

// ---------------------------------------------------------------------------
// phi = elu(x)+1 applied in-place to q,k region of qkv [16384, 3072]
// (columns [0, 2048); elu(x)+1 = x>0 ? x+1 : exp(x))
// ---------------------------------------------------------------------------
__global__ __launch_bounds__(256) void phi_kernel(float* __restrict__ qkv) {
  size_t idx = ((size_t)blockIdx.x * 256 + threadIdx.x) * 4;   // over 16384*2048
  if (idx >= (size_t)16384 * 2048) return;
  size_t row = idx >> 11;
  int    col = (int)(idx & 2047);
  float* p = qkv + row * 3072 + col;
  float4 v = *(float4*)p;
  v.x = v.x > 0.0f ? v.x + 1.0f : __expf(v.x);
  v.y = v.y > 0.0f ? v.y + 1.0f : __expf(v.y);
  v.z = v.z > 0.0f ? v.z + 1.0f : __expf(v.z);
  v.w = v.w > 0.0f ? v.w + 1.0f : __expf(v.w);
  *(float4*)p = v;
}

// ---------------------------------------------------------------------------
// kv[b,h,e,f] = sum_n phi_k[n,e] * v[n,f];  ksum[b,h,e] = sum_n phi_k[n,e]
// One block per (b,h).  qkv layout: row*3072 + {q:0..1023, k:1024.., v:2048..},
// channel = h*64 + e.
// ---------------------------------------------------------------------------
__global__ __launch_bounds__(256) void kv_ksum_kernel(
    const float* __restrict__ qkv, float* __restrict__ kvb,
    float* __restrict__ ksum) {
  const int bh = blockIdx.x;
  const int b = bh >> 4, h = bh & 15;
  const int t = threadIdx.x;
  __shared__ float ks[32][68];
  __shared__ float vs[32][68];
  const int e = t & 63, fg = t >> 6;           // fg in 0..3 -> 16 f's each
  float acc[16];
#pragma unroll
  for (int i = 0; i < 16; ++i) acc[i] = 0.0f;
  float ksacc = 0.0f;
  const int lr = t >> 3, lc = (t & 7) * 8;     // 32 rows x 64 cols, 8 floats/thr
  const size_t rb = ((size_t)b * 4096) * 3072 + (size_t)h * 64;
  for (int n0 = 0; n0 < 4096; n0 += 32) {
    const float* kp = qkv + rb + (size_t)(n0 + lr) * 3072 + 1024 + lc;
    const float* vp = qkv + rb + (size_t)(n0 + lr) * 3072 + 2048 + lc;
    float4 k0 = ((const float4*)kp)[0], k1 = ((const float4*)kp)[1];
    float4 v0 = ((const float4*)vp)[0], v1 = ((const float4*)vp)[1];
    __syncthreads();
    *(float4*)&ks[lr][lc]     = k0;  *(float4*)&ks[lr][lc + 4] = k1;
    *(float4*)&vs[lr][lc]     = v0;  *(float4*)&vs[lr][lc + 4] = v1;
    __syncthreads();
#pragma unroll 4
    for (int j = 0; j < 32; ++j) {
      float kj = ks[j][e];
      if (fg == 0) ksacc += kj;
#pragma unroll
      for (int ff = 0; ff < 16; ++ff) acc[ff] += kj * vs[j][fg * 16 + ff];
    }
  }
  float* o = kvb + (size_t)bh * 4096 + (size_t)e * 64 + fg * 16;
#pragma unroll
  for (int ff = 0; ff < 16; ++ff) o[ff] = acc[ff];
  if (fg == 0) ksum[(size_t)bh * 64 + e] = ksacc;
}

// ---------------------------------------------------------------------------
// attn_out[b,n,h*64+f] = z * sum_e q[n,e]*kv[e,f];  z = 1/(q . ksum + eps)
// One block per (b,h, 64-token chunk).  Output written as bf16 (GEMM A input).
// ---------------------------------------------------------------------------
__global__ __launch_bounds__(256) void attn_kernel(
    const float* __restrict__ qkv, const float* __restrict__ kvb,
    const float* __restrict__ ksum, unsigned short* __restrict__ attnA) {
  const int bh = blockIdx.x;
  const int b = bh >> 4, h = bh & 15;
  const int n0 = blockIdx.y * 64;
  const int t = threadIdx.x;
  __shared__ float kvs[64][68];
  __shared__ float qs[64][68];
  __shared__ float kss[64];
  const int lr = t >> 2, lc = (t & 3) * 16;    // 64 rows x 64 cols, 16 f/thr
  {
    const float4* kvp = (const float4*)(kvb + (size_t)bh * 4096 + (size_t)lr * 64 + lc);
    float4 a0 = kvp[0], a1 = kvp[1], a2 = kvp[2], a3 = kvp[3];
    const float4* qp = (const float4*)(qkv + (size_t)(b * 4096 + n0 + lr) * 3072 + h * 64 + lc);
    float4 q0 = qp[0], q1 = qp[1], q2 = qp[2], q3 = qp[3];
    *(float4*)&kvs[lr][lc]      = a0; *(float4*)&kvs[lr][lc + 4]  = a1;
    *(float4*)&kvs[lr][lc + 8]  = a2; *(float4*)&kvs[lr][lc + 12] = a3;
    *(float4*)&qs[lr][lc]       = q0; *(float4*)&qs[lr][lc + 4]   = q1;
    *(float4*)&qs[lr][lc + 8]   = q2; *(float4*)&qs[lr][lc + 12]  = q3;
    if (t < 64) kss[t] = ksum[(size_t)bh * 64 + t];
  }
  __syncthreads();
  const int i = t & 63, fg = t >> 6;
  float zd = 0.0f;
#pragma unroll 8
  for (int e = 0; e < 64; ++e) zd += qs[i][e] * kss[e];
  float z = 1.0f / (zd + 1e-6f);
  float acc[16];
#pragma unroll
  for (int ff = 0; ff < 16; ++ff) acc[ff] = 0.0f;
#pragma unroll 4
  for (int e = 0; e < 64; ++e) {
    float qv = qs[i][e];
#pragma unroll
    for (int ff = 0; ff < 16; ++ff) acc[ff] += qv * kvs[e][fg * 16 + ff];
  }
  unsigned short* op = attnA + (size_t)(b * 4096 + n0 + i) * 1024 + h * 64 + fg * 16;
#pragma unroll
  for (int ff = 0; ff < 16; ++ff) op[ff] = f2bf(acc[ff] * z);
}

// ---------------------------------------------------------------------------
// bf16 WMMA GEMM:  D[M,Nout] = A[M,K] @ W[Nout,K]^T  (torch Linear layout)
// Block tile 128x128, 8 waves of 64x32, BK=32 -> one v_wmma_f32_16x16x32_bf16
// per 16x16 tile per K-step.  W tile fetched by the Tensor Data Mover into
// LDS with D# row padding matching the +8-ushort bank-conflict pad.
// EPI: 0 = store f32; 1 = f32 + bias + residual; 2 = bf16 gelu(acc+bias).
// ---------------------------------------------------------------------------
template <int EPI>
__global__ __launch_bounds__(256) void gemm_bf16_wmma(
    const unsigned short* __restrict__ A, const unsigned short* __restrict__ W,
    const float* __restrict__ bias, const float* __restrict__ resid,
    void* __restrict__ Dout, int M, int Nout, int K) {
  __shared__ unsigned short As[128][40];   // 32 + 8 pad (80B stride)
  __shared__ unsigned short Bs[128][40];
  const int tid  = threadIdx.x;
  const int lane = tid & 31, wave = tid >> 5;
  const int bm = blockIdx.y * 128, bn = blockIdx.x * 128;
  const int wm = (wave & 1) * 64, wn = (wave >> 1) * 32;

  v8f acc[4][2] = {};

  const int lrow = tid >> 1, lk = (tid & 1) * 16;
  const unsigned short* Ag = A + (size_t)(bm + lrow) * K + lk;
#if !USE_TDM
  const unsigned short* Wg = W + (size_t)(bn + lrow) * K + lk;
#endif

  for (int k0 = 0; k0 < K; k0 += 32) {
    uint4 a0 = *(const uint4*)(Ag + k0);
    uint4 a1 = *(const uint4*)(Ag + k0 + 8);
#if !USE_TDM
    uint4 b0 = *(const uint4*)(Wg + k0);
    uint4 b1 = *(const uint4*)(Wg + k0 + 8);
#endif
    __syncthreads();   // previous iteration's compute done reading LDS
    *(uint4*)&As[lrow][lk]     = a0;
    *(uint4*)&As[lrow][lk + 8] = a1;
#if USE_TDM
    if (wave == 0) {
      // ---- Tensor DMA descriptor: 2D tile 32(k) x 128(n) bf16 of W ----
      unsigned long long ga =
          (unsigned long long)(uintptr_t)W +
          ((unsigned long long)(unsigned)bn * (unsigned)K + (unsigned)k0) * 2ull;
      unsigned int lds = (unsigned int)(uintptr_t)&Bs[0][0];
      v4u g0;
      g0[0] = 1u;                                     // count=1 (valid D#)
      g0[1] = lds;                                    // lds_addr [63:32]
      g0[2] = (unsigned)(ga & 0xFFFFFFFFull);         // global_addr [95:64]
      g0[3] = (unsigned)((ga >> 32) & 0x1FFFFFFull)   // global_addr [120:96]
              | (2u << 30);                           // type=2 (image)
      v8i g1;
      g1[0] = (int)((1u << 16)      // data_size = 1 (2 bytes)
                  | (1u << 20)      // pad_enable
                  | (3u << 22)      // pad_interval: 16 DWORDs (one 64B row)
                  | (3u << 25));    // pad_amount: 4 DWORDs (16B) -> stride 80B
      g1[1] = (int)(((unsigned)K & 0xFFFFu) << 16);                       // tensor_dim0 lo
      g1[2] = (int)((((unsigned)K >> 16) & 0xFFFFu)
                  | (((unsigned)Nout & 0xFFFFu) << 16));                  // dim0 hi | dim1 lo
      g1[3] = (int)((((unsigned)Nout >> 16) & 0xFFFFu) | (32u << 16));    // dim1 hi | tile_dim0=32
      g1[4] = 128;                                                        // tile_dim1 = 128
      g1[5] = K;                                                          // tensor_dim0_stride lo
      g1[6] = 0;
      g1[7] = 0;
      v4i gz4; gz4[0] = gz4[1] = gz4[2] = gz4[3] = 0;
      v8i gz8;
      gz8[0] = gz8[1] = gz8[2] = gz8[3] = 0;
      gz8[4] = gz8[5] = gz8[6] = gz8[7] = 0;
      // 6-arg form (clang-23 / therock-10.0 headers): groups 0..3 + spare + cpol
      __builtin_amdgcn_tensor_load_to_lds(g0, g1, gz4, gz4, gz8, 0);
      __builtin_amdgcn_s_wait_tensorcnt(0);
    }
#else
    *(uint4*)&Bs[lrow][lk]     = b0;
    *(uint4*)&Bs[lrow][lk + 8] = b1;
#endif
    __syncthreads();   // tiles visible to all waves

    // ---- fragments (ISA 7.12.2 layouts) ----
    union FA { uint4 u[2]; v16bf v; };
    const int half = lane >> 4;
    FA af[4];
#pragma unroll
    for (int mt = 0; mt < 4; ++mt) {
      const unsigned short* ap = &As[wm + mt * 16 + (lane & 15)][half * 8];
      af[mt].u[0] = *(const uint4*)ap;          // k = half*8 + 0..7
      af[mt].u[1] = *(const uint4*)(ap + 16);   // k = 16 + half*8 + 0..7
    }
    FA bfr[2];
    const int kb = half * 8;
#pragma unroll
    for (int nt = 0; nt < 2; ++nt) {
      const unsigned short* bp = &Bs[wn + nt * 16 + (lane & 15)][kb];
      bfr[nt].u[0] = *(const uint4*)bp;         // k = kb + 0..7
      bfr[nt].u[1] = *(const uint4*)(bp + 16);  // k = 16 + kb + 0..7
    }
#pragma unroll
    for (int mt = 0; mt < 4; ++mt)
#pragma unroll
      for (int nt = 0; nt < 2; ++nt)
        acc[mt][nt] = __builtin_amdgcn_wmma_f32_16x16x32_bf16(
            false, af[mt].v, false, bfr[nt].v, (short)0, acc[mt][nt],
            false, false);
  }

  // ---- epilogue: C/D layout row = 8*(lane/16)+i, col = lane%16 ----
#pragma unroll
  for (int mt = 0; mt < 4; ++mt) {
#pragma unroll
    for (int nt = 0; nt < 2; ++nt) {
#pragma unroll
      for (int i = 0; i < 8; ++i) {
        int row = bm + wm + mt * 16 + (lane >> 4) * 8 + i;
        int col = bn + wn + nt * 16 + (lane & 15);
        float v = acc[mt][nt][i];
        if (EPI == 0) {
          ((float*)Dout)[(size_t)row * Nout + col] = v;
        } else if (EPI == 1) {
          v += bias[col] + resid[(size_t)row * Nout + col];
          ((float*)Dout)[(size_t)row * Nout + col] = v;
        } else {
          v += bias[col];
          float gl = 0.5f * v * (1.0f + erff(v * 0.70710678118654752f));
          ((unsigned short*)Dout)[(size_t)row * Nout + col] = f2bf(gl);
        }
      }
    }
  }
}

// ---------------------------------------------------------------------------
extern "C" void kernel_launch(void* const* d_in, const int* in_sizes, int n_in,
                              void* d_out, int out_size, void* d_ws, size_t ws_size,
                              hipStream_t stream) {
  const float* x      = (const float*)d_in[0];
  const float* n1g    = (const float*)d_in[1];
  const float* n1b    = (const float*)d_in[2];
  const float* qkv_w  = (const float*)d_in[3];
  const float* proj_w = (const float*)d_in[4];
  const float* proj_b = (const float*)d_in[5];
  const float* n2g    = (const float*)d_in[6];
  const float* n2b    = (const float*)d_in[7];
  const float* fc1_w  = (const float*)d_in[8];
  const float* fc1_b  = (const float*)d_in[9];
  const float* fc2_w  = (const float*)d_in[10];
  const float* fc2_b  = (const float*)d_in[11];

  const size_t Mrows = 16384;   // B*N
  char* ws = (char*)d_ws;
  size_t off = 0;
  auto alloc = [&](size_t bytes) -> char* {
    char* p = ws + off;
    off += (bytes + 255) & ~(size_t)255;
    return p;
  };
  unsigned short* hbuf  = (unsigned short*)alloc(Mrows * 1024 * 2);  // LN out (reused for LN2)
  unsigned short* wqkv  = (unsigned short*)alloc((size_t)3072 * 1024 * 2);
  unsigned short* wproj = (unsigned short*)alloc((size_t)1024 * 1024 * 2);
  unsigned short* wfc1  = (unsigned short*)alloc((size_t)4096 * 1024 * 2);
  unsigned short* wfc2  = (unsigned short*)alloc((size_t)1024 * 4096 * 2);
  float*          qkv   = (float*)alloc(Mrows * 3072 * 4);           // reused for fc1 act
  float*          kvb   = (float*)alloc((size_t)64 * 4096 * 4);
  float*          ksum  = (float*)alloc((size_t)64 * 64 * 4);
  unsigned short* attnA = (unsigned short*)alloc(Mrows * 1024 * 2);
  float*          x1    = (float*)alloc(Mrows * 1024 * 4);
  unsigned short* h2    = hbuf;                 // reuse: LN1 out dead after qkv GEMM
  unsigned short* h3    = (unsigned short*)qkv; // reuse: qkv dead after attn kernel

  dim3 blk(256);

  // weights -> bf16
  f32_to_bf16_kernel<<<3072, blk, 0, stream>>>(qkv_w,  wqkv,  3072 * 1024);
  f32_to_bf16_kernel<<<1024, blk, 0, stream>>>(proj_w, wproj, 1024 * 1024);
  f32_to_bf16_kernel<<<4096, blk, 0, stream>>>(fc1_w,  wfc1,  4096 * 1024);
  f32_to_bf16_kernel<<<4096, blk, 0, stream>>>(fc2_w,  wfc2,  1024 * 4096);

  // LN1 -> bf16
  ln_to_bf16_kernel<<<16384, blk, 0, stream>>>(x, n1g, n1b, hbuf);

  // qkv = h @ qkv_w^T   [16384 x 3072]
  gemm_bf16_wmma<0><<<dim3(24, 128), blk, 0, stream>>>(
      hbuf, wqkv, nullptr, nullptr, qkv, 16384, 3072, 1024);

  // phi on q,k
  phi_kernel<<<32768, blk, 0, stream>>>(qkv);

  // kv / ksum per head
  kv_ksum_kernel<<<64, blk, 0, stream>>>(qkv, kvb, ksum);

  // attention apply -> bf16
  attn_kernel<<<dim3(64, 64), blk, 0, stream>>>(qkv, kvb, ksum, attnA);

  // x1 = x + attn @ proj_w^T + proj_b
  gemm_bf16_wmma<1><<<dim3(8, 128), blk, 0, stream>>>(
      attnA, wproj, proj_b, x, x1, 16384, 1024, 1024);

  // LN2 -> bf16
  ln_to_bf16_kernel<<<16384, blk, 0, stream>>>(x1, n2g, n2b, h2);

  // h3 = gelu(h2 @ fc1_w^T + fc1_b)  -> bf16
  gemm_bf16_wmma<2><<<dim3(32, 128), blk, 0, stream>>>(
      h2, wfc1, fc1_b, nullptr, h3, 16384, 4096, 1024);

  // out = x1 + h3 @ fc2_w^T + fc2_b
  gemm_bf16_wmma<1><<<dim3(8, 128), blk, 0, stream>>>(
      h3, wfc2, fc2_b, x1, (float*)d_out, 16384, 1024, 4096);
}
